// SelfAttention_82781199663494
// MI455X (gfx1250) — matrix operand order repaired and checked by reading, not compiled, not run on hardware
//
#include <hip/hip_runtime.h>

// ---------------------------------------------------------------------------
// Self-attention (B=8, C=64, H=W=64 -> N=4096, CQ=8) for gfx1250 (MI455X).
// Flash-attention structure, f16 WMMA (v_wmma_f32_16x16x32_f16), f32 softmax.
// V tiles staged through LDS with async global->LDS copies (ASYNCcnt),
// double-buffered across the j-loop, shared by the 8 waves of a block.
// ---------------------------------------------------------------------------

typedef __attribute__((ext_vector_type(16))) _Float16 v16h;
typedef __attribute__((ext_vector_type(8)))  _Float16 v8h;
typedef __attribute__((ext_vector_type(8)))  float    v8f;

// pointee type for the async-LDS builtin, exactly as clang's diagnostic spells it
typedef int vs4i __attribute__((vector_size(4 * sizeof(int))));
typedef __attribute__((address_space(1))) vs4i* gptr4i;
typedef __attribute__((address_space(3))) vs4i* lptr4i;

#define WMMA_F16(A_, B_, C_) \
  __builtin_amdgcn_wmma_f32_16x16x32_f16(false, (A_), false, (B_), (short)0, (C_), false, false)

constexpr int kB  = 8;
constexpr int kC  = 64;
constexpr int kN  = 4096;   // H*W
constexpr int kCQ = 8;
constexpr int kVROW = 40;   // padded LDS row stride (halfs): 80B -> conflict-free b128 reads

#if defined(__HIP_DEVICE_COMPILE__) && __has_builtin(__builtin_amdgcn_global_load_async_to_lds_b128)
#define HAVE_ASYNC_LDS 1
#else
#define HAVE_ASYNC_LDS 0
#endif

// ---------------------------------------------------------------------------
// Kernel 1: q/k/v projections as WMMA GEMMs.
// One wave per (batch, 16-pixel) tile; K=64 split in two K=32 WMMA chunks,
// bias folded into initial C. q16,k16: [B][N][8] f16 ; v16: [B][C][N] f16.
// ---------------------------------------------------------------------------
__global__ __launch_bounds__(256) void qkv_kernel(
    const float* __restrict__ x,
    const float* __restrict__ wq, const float* __restrict__ bq,
    const float* __restrict__ wk, const float* __restrict__ bk,
    const float* __restrict__ wv, const float* __restrict__ bv,
    _Float16* __restrict__ q16, _Float16* __restrict__ k16,
    _Float16* __restrict__ v16)
{
  const int tid  = blockIdx.x * blockDim.x + threadIdx.x;
  const int wave = tid >> 5;
  const int lane = threadIdx.x & 31;
  const int lm   = lane & 15;
  const int hg   = lane >> 4;

  const int b  = wave >> 8;
  const int n0 = (wave & 255) << 4;

  // B operand: x columns, two K=32 chunks
  v16h bx[2];
#pragma unroll
  for (int ch = 0; ch < 2; ++ch) {
#pragma unroll
    for (int h = 0; h < 16; ++h) {
      const int c = ch * 32 + (h >> 3) * 16 + hg * 8 + (h & 7);
      bx[ch][h] = (_Float16)x[(size_t)(b * kC + c) * kN + n0 + lm];
    }
  }

  // A operand for fused q/k: rows 0..7 = wq, rows 8..15 = wk
  const float* wrow = (lm < 8) ? (wq + lm * kC) : (wk + (lm - 8) * kC);
  v16h aqk[2];
#pragma unroll
  for (int ch = 0; ch < 2; ++ch)
#pragma unroll
    for (int g = 0; g < 2; ++g)
#pragma unroll
      for (int o = 0; o < 8; ++o)
        aqk[ch][g * 8 + o] = (_Float16)wrow[ch * 32 + g * 16 + hg * 8 + o];

  v8f dqk;
#pragma unroll
  for (int r = 0; r < 8; ++r) dqk[r] = hg ? bk[r] : bq[r];
  dqk = WMMA_F16(aqk[0], bx[0], dqk);
  dqk = WMMA_F16(aqk[1], bx[1], dqk);

  v8h hrow;
#pragma unroll
  for (int r = 0; r < 8; ++r) hrow[r] = (_Float16)dqk[r];
  _Float16* qkdst = (hg ? k16 : q16) + (size_t)(b * kN + n0 + lm) * kCQ;
  *(v8h*)qkdst = hrow;

#pragma unroll
  for (int t = 0; t < 4; ++t) {
    const float* wvrow = wv + (16 * t + lm) * kC;
    v16h av[2];
#pragma unroll
    for (int ch = 0; ch < 2; ++ch)
#pragma unroll
      for (int g = 0; g < 2; ++g)
#pragma unroll
        for (int o = 0; o < 8; ++o)
          av[ch][g * 8 + o] = (_Float16)wvrow[ch * 32 + g * 16 + hg * 8 + o];

    v8f dv;
#pragma unroll
    for (int r = 0; r < 8; ++r) dv[r] = bv[16 * t + hg * 8 + r];
    dv = WMMA_F16(av[0], bx[0], dv);
    dv = WMMA_F16(av[1], bx[1], dv);

#pragma unroll
    for (int r = 0; r < 8; ++r) {
      const int co = 16 * t + hg * 8 + r;
      v16[(size_t)(b * kC + co) * kN + n0 + lm] = (_Float16)dv[r];
    }
  }
}

// ---------------------------------------------------------------------------
// Kernel 2: flash attention with async-LDS staged V.
// One wave owns 2 query tiles (32 queries); the 8 waves of a block share the
// same batch and j-loop, so V[64c x 32j] is staged once per block in LDS.
// ---------------------------------------------------------------------------
__global__ __launch_bounds__(256) void attn_kernel(
    const _Float16* __restrict__ q16,
    const _Float16* __restrict__ k16,
    const _Float16* __restrict__ v16,
    float* __restrict__ out)
{
  __shared__ _Float16 vlds[2][kC * kVROW];   // 2 x 5120B, double-buffered

  const int tid  = blockIdx.x * blockDim.x + threadIdx.x;
  const int wave = tid >> 5;
  const int lane = threadIdx.x & 31;
  const int lm   = lane & 15;
  const int hg   = lane >> 4;
  const int ltid = threadIdx.x;

  const int b  = wave >> 7;          // 8 waves/block all share b (8 | 128)
  const int i0 = (wave & 127) << 5;

  // per-thread slice of the cooperative V stage: one b128 per thread
  const int vrow = ltid >> 2;               // 0..63  (channel c)
  const int vjj  = (ltid & 3) * 8;          // 0/8/16/24 (j within block)
  const _Float16* vsrc_base = v16 + (size_t)(b * kC + vrow) * kN + vjj;

  // ---- B operand: Q^T per i-tile (K-dim = channel, 8 real + zero pad) -----
  v16h bqop[2];
#pragma unroll
  for (int it = 0; it < 2; ++it)
#pragma unroll
    for (int h = 0; h < 16; ++h) bqop[it][h] = (_Float16)0.f;
  if (hg == 0) {
#pragma unroll
    for (int it = 0; it < 2; ++it) {
      const v8h qr = *(const v8h*)(q16 + (size_t)(b * kN + i0 + 16 * it + lm) * kCQ);
#pragma unroll
      for (int h = 0; h < 8; ++h) bqop[it][h] = qr[h];
    }
  }

  float m_[2] = {-1e30f, -1e30f};
  float l_[2] = {0.f, 0.f};
  v8f Oacc[2][4];
#pragma unroll
  for (int it = 0; it < 2; ++it)
#pragma unroll
    for (int t = 0; t < 4; ++t)
#pragma unroll
      for (int r = 0; r < 8; ++r) Oacc[it][t][r] = 0.f;

  v8f zc;
#pragma unroll
  for (int r = 0; r < 8; ++r) zc[r] = 0.f;

  // ---- issue the cooperative async V stage for one j-block ----------------
  auto issue_vstage = [&](int buf, int j0) {
    _Float16* ldst = &vlds[buf][vrow * kVROW + vjj];
#if HAVE_ASYNC_LDS
    __builtin_amdgcn_global_load_async_to_lds_b128(
        (gptr4i)(vsrc_base + j0), (lptr4i)ldst, 0, 0);
#else
    *(v8h*)ldst = *(const v8h*)(vsrc_base + j0);
#endif
  };

  issue_vstage(0, 0);                        // prologue: stage first block

  for (int jb = 0; jb < kN / 32; ++jb) {
    const int j0  = jb * 32;
    const int buf = jb & 1;

    __syncthreads();                         // readers of buf^1 are done
    if (jb + 1 < kN / 32) {
      issue_vstage(buf ^ 1, j0 + 32);        // overlap next stage with compute
#if HAVE_ASYNC_LDS
#if __has_builtin(__builtin_amdgcn_s_wait_asynccnt)
      __builtin_amdgcn_s_wait_asynccnt(1);   // current buffer landed
#else
      asm volatile("s_wait_asynccnt 0x1" ::: "memory");
#endif
#endif
    } else {
#if HAVE_ASYNC_LDS
#if __has_builtin(__builtin_amdgcn_s_wait_asynccnt)
      __builtin_amdgcn_s_wait_asynccnt(0);
#else
      asm volatile("s_wait_asynccnt 0x0" ::: "memory");
#endif
#endif
    }
    __syncthreads();                         // all waves' stage complete

    // ---- K-tile A operands (rows j, K-dim = channel, zero-padded) ---------
    v16h ak0, ak1;
#pragma unroll
    for (int h = 0; h < 16; ++h) { ak0[h] = (_Float16)0.f; ak1[h] = (_Float16)0.f; }
    if (hg == 0) {
      const v8h k0 = *(const v8h*)(k16 + (size_t)(b * kN + j0 + lm) * kCQ);
      const v8h k1 = *(const v8h*)(k16 + (size_t)(b * kN + j0 + 16 + lm) * kCQ);
#pragma unroll
      for (int h = 0; h < 8; ++h) { ak0[h] = k0[h]; ak1[h] = k1[h]; }
      if (j0 + 32 < kN)
        __builtin_prefetch(k16 + (size_t)(b * kN + j0 + 32 + lm) * kCQ, 0, 1);
    }

    v16h bp[2];
#pragma unroll
    for (int it = 0; it < 2; ++it) {
      // S^T tiles: lane holds column i; pair (L, L+16) covers all 32 j
      v8f s0 = WMMA_F16(ak0, bqop[it], zc);
      v8f s1 = WMMA_F16(ak1, bqop[it], zc);

      float bm = s0[0];
#pragma unroll
      for (int r = 1; r < 8; ++r) bm = fmaxf(bm, s0[r]);
#pragma unroll
      for (int r = 0; r < 8; ++r) bm = fmaxf(bm, s1[r]);
      bm = fmaxf(bm, __shfl_xor(bm, 16));

      const float mn    = fmaxf(m_[it], bm);
      const float alpha = __expf(m_[it] - mn);

      float p0f[8], p1f[8];
      float bs = 0.f;
#pragma unroll
      for (int r = 0; r < 8; ++r) {
        p0f[r] = __expf(s0[r] - mn);
        p1f[r] = __expf(s1[r] - mn);
        bs += p0f[r] + p1f[r];
      }
      bs += __shfl_xor(bs, 16);

      l_[it] = l_[it] * alpha + bs;
      m_[it] = mn;
#pragma unroll
      for (int t = 0; t < 4; ++t)
#pragma unroll
        for (int r = 0; r < 8; ++r) Oacc[it][t][r] *= alpha;

      // transpose P^T into B-operand layout via xor-16 lane exchange
#pragma unroll
      for (int r = 0; r < 8; ++r) {
        const float pp0 = __shfl_xor(p0f[r], 16);
        const float pp1 = __shfl_xor(p1f[r], 16);
        const float lo  = hg ? pp1 : p0f[r];
        const float hi  = hg ? p1f[r] : pp0;
        bp[it][r]     = (_Float16)lo;
        bp[it][8 + r] = (_Float16)hi;
      }
    }

    // ---- GEMM2: O[c,i] += V[c, j-block] * P^T, V read from LDS ------------
#pragma unroll
    for (int t = 0; t < 4; ++t) {
      const _Float16* vb = &vlds[buf][(16 * t + lm) * kVROW + hg * 16];
      const v8h vlo = *(const v8h*)(vb);
      const v8h vhi = *(const v8h*)(vb + 8);
      v16h av;
#pragma unroll
      for (int h = 0; h < 8; ++h) { av[h] = vlo[h]; av[8 + h] = vhi[h]; }
#pragma unroll
      for (int it = 0; it < 2; ++it)
        Oacc[it][t] = WMMA_F16(av, bp[it], Oacc[it][t]);
    }
  }

  // ---- epilogue -----------------------------------------------------------
#pragma unroll
  for (int it = 0; it < 2; ++it) {
    const float inv = 1.0f / l_[it];
    const int   i   = i0 + 16 * it + lm;
#pragma unroll
    for (int t = 0; t < 4; ++t)
#pragma unroll
      for (int r = 0; r < 8; ++r) {
        const int c = 16 * t + hg * 8 + r;
        out[(size_t)(b * kC + c) * kN + i] = Oacc[it][t][r] * inv;
      }
  }
}

// ---------------------------------------------------------------------------
extern "C" void kernel_launch(void* const* d_in, const int* in_sizes, int n_in,
                              void* d_out, int out_size, void* d_ws, size_t ws_size,
                              hipStream_t stream) {
  const float* x  = (const float*)d_in[0];
  const float* wq = (const float*)d_in[1];
  const float* bq = (const float*)d_in[2];
  const float* wk = (const float*)d_in[3];
  const float* bk = (const float*)d_in[4];
  const float* wv = (const float*)d_in[5];
  const float* bv = (const float*)d_in[6];
  float* out = (float*)d_out;

  _Float16* q16 = (_Float16*)d_ws;
  _Float16* k16 = q16 + (size_t)kB * kN * kCQ;
  _Float16* v16 = k16 + (size_t)kB * kN * kCQ;

  qkv_kernel<<<256, 256, 0, stream>>>(x, wq, bq, wk, bk, wv, bv, q16, k16, v16);
  attn_kernel<<<128, 256, 0, stream>>>(q16, k16, v16, out);
}
